// CaptioningRNN_80702435492284
// MI455X (gfx1250) — matrix operand order repaired and checked by reading, not compiled, not run on hardware
//
#include <hip/hip_runtime.h>

#define N_BATCH 1024
#define T_STEPS 64
#define D_IN    512
#define H_DIM   1024
#define FH      4096      // 4*H
#define K_TOT   2560      // D + 2H   ([xt | h | attn] concatenated K)
#define KC      32        // K-chunk per WMMA
#define NCHUNK  (K_TOT / KC)   // 80

typedef __attribute__((ext_vector_type(16))) __bf16 bf16x16;
typedef __attribute__((ext_vector_type(8)))  __bf16 bf16x8;
typedef __attribute__((ext_vector_type(8)))  float  f32x8;

#if defined(__HIP_DEVICE_COMPILE__) && \
    __has_builtin(__builtin_amdgcn_tensor_load_to_lds) && \
    __has_builtin(__builtin_amdgcn_s_wait_tensorcnt)
#define HAVE_TDM 1
#else
#define HAVE_TDM 0
#endif

__device__ __forceinline__ unsigned short f2bf(float f) {
    union { float f; unsigned u; } v; v.f = f;
    unsigned u = v.u;
    u += 0x7FFFu + ((u >> 16) & 1u);     // round-to-nearest-even
    return (unsigned short)(u >> 16);
}

__device__ __forceinline__ float sigmoidf_(float x) {
    return 1.0f / (1.0f + __expf(-x));
}

#if HAVE_TDM
typedef unsigned int u32x4v __attribute__((ext_vector_type(4)));
typedef int          i32x4v __attribute__((ext_vector_type(4)));
typedef int          i32x8v __attribute__((ext_vector_type(8)));

// 2D tensor-tile DMA: global (row-major, stride0 elements of 2B) -> LDS,
// per CDNA5 D# layout (CDNA5 ISA). data_size = 2 bytes.
__device__ __forceinline__ void tdm_load_2d(const void* gptr, unsigned ldsOff,
                                            unsigned td0, unsigned td1,
                                            unsigned tile0, unsigned tile1,
                                            unsigned stride0) {
    unsigned long long ga = (unsigned long long)(size_t)gptr;
    u32x4v g0;
    g0[0] = 1u;                                    // count=1 (valid user D#)
    g0[1] = ldsOff;                                // lds_addr (bytes)
    g0[2] = (unsigned)(ga & 0xFFFFFFFFu);          // global_addr[31:0]
    g0[3] = (unsigned)((ga >> 32) & 0x01FFFFFFu)   // global_addr[56:32]
          | 0x80000000u;                           // type = 2 ("image")
    i32x8v g1;
    g1[0] = (int)(1u << 16);                       // data_size=1 (2B); mask=0
    g1[1] = (int)((td0 & 0xFFFFu) << 16);                          // tensor_dim0 lo
    g1[2] = (int)(((td0 >> 16) & 0xFFFFu) | ((td1 & 0xFFFFu) << 16));
    g1[3] = (int)(((td1 >> 16) & 0xFFFFu) | (tile0 << 16));        // tile_dim0
    g1[4] = (int)(tile1 & 0xFFFFu);                                // tile_dim1 (tile_dim2=0)
    g1[5] = (int)stride0;                                          // tensor_dim0_stride lo32
    g1[6] = 0;                                                     // stride hi / dim1_stride
    g1[7] = 0;
    i32x4v z = {0, 0, 0, 0};
#if __clang_major__ >= 23
    i32x8v z8 = {0, 0, 0, 0, 0, 0, 0, 0};
    __builtin_amdgcn_tensor_load_to_lds(g0, g1, z, z, z8, 0);
#else
    __builtin_amdgcn_tensor_load_to_lds(g0, g1, z, z, 0);
#endif
}
#endif // HAVE_TDM

// ---------------------------------------------------------------------------
// Gate-interleaved packed weights: WcatT[pc][k], pc = j*4 + gate, k in [0,2560)
// so one block's 256 needed columns are CONTIGUOUS rows -> single TDM tile.
// ---------------------------------------------------------------------------
__global__ void pack_weights(const float* __restrict__ Wx,
                             const float* __restrict__ Wh,
                             const float* __restrict__ Wa,
                             unsigned short* __restrict__ WcatT) {
    int pc = blockIdx.x * blockDim.x + threadIdx.x;    // 0..4095
    if (pc >= FH) return;
    int j    = pc >> 2;
    int gate = pc & 3;
    int col  = gate * H_DIM + j;                       // original 4H column
    unsigned short* dst = WcatT + (size_t)pc * K_TOT;
    for (int k = 0; k < D_IN; ++k)
        dst[k] = f2bf(Wx[(size_t)k * FH + col]);
    for (int k = 0; k < H_DIM; ++k)
        dst[D_IN + k] = f2bf(Wh[(size_t)k * FH + col]);
    for (int k = 0; k < H_DIM; ++k)
        dst[D_IN + H_DIM + k] = f2bf(Wa[(size_t)k * FH + col]);
}

// ---------------------------------------------------------------------------
// h0 = c0 = mean over 4x4 grid of A; seed fp32 state and bf16 h-columns of Acat
// ---------------------------------------------------------------------------
__global__ void init_state(const float* __restrict__ A,
                           float* __restrict__ c_st,
                           float* __restrict__ h_st,
                           unsigned short* __restrict__ Acat) {
    size_t idx = (size_t)blockIdx.x * blockDim.x + threadIdx.x;   // n*H + j
    const float4* p = reinterpret_cast<const float4*>(A + idx * 16);
    float4 q0 = p[0], q1 = p[1], q2 = p[2], q3 = p[3];
    float s = q0.x + q0.y + q0.z + q0.w + q1.x + q1.y + q1.z + q1.w
            + q2.x + q2.y + q2.z + q2.w + q3.x + q3.y + q3.z + q3.w;
    float h0 = s * (1.0f / 16.0f);
    c_st[idx] = h0;
    h_st[idx] = h0;
    size_t n = idx >> 10, j = idx & 1023;
    Acat[n * K_TOT + D_IN + j] = f2bf(h0);
}

// ---------------------------------------------------------------------------
// Per-step attention (fp32) + x[:,t,:] bf16 conversion into Acat.
// ---------------------------------------------------------------------------
__global__ void __launch_bounds__(256)
attn_step(const float* __restrict__ x,        // (N,T,D)
          const float* __restrict__ A,        // (N,H,16)
          const float* __restrict__ h_st,     // (N,H) fp32
          unsigned short* __restrict__ Acat,  // (N,K_TOT) bf16
          int t) {
    __shared__ float red[256 * 16];
    const int n   = blockIdx.x;
    const int tid = threadIdx.x;

    const float* Af = A    + (size_t)n * H_DIM * 16;
    const float* hv = h_st + (size_t)n * H_DIM;

    float av[4][16];
    float s[16];
#pragma unroll
    for (int l = 0; l < 16; ++l) s[l] = 0.0f;

    const int j0 = tid * 4;
#pragma unroll
    for (int jj = 0; jj < 4; ++jj) {
        int j = j0 + jj;
        float hr = hv[j];
        const float4* p = reinterpret_cast<const float4*>(Af + (size_t)j * 16);
        float4* dstq = reinterpret_cast<float4*>(av[jj]);
        dstq[0] = p[0]; dstq[1] = p[1]; dstq[2] = p[2]; dstq[3] = p[3];
#pragma unroll
        for (int l = 0; l < 16; ++l) s[l] += hr * av[jj][l];
    }

#pragma unroll
    for (int l = 0; l < 16; ++l) red[tid * 16 + l] = s[l];
    __syncthreads();
    for (int stride = 128; stride > 0; stride >>= 1) {
        if (tid < stride) {
#pragma unroll
            for (int l = 0; l < 16; ++l)
                red[tid * 16 + l] += red[(tid + stride) * 16 + l];
        }
        __syncthreads();
    }

    const float scale = 0.03125f;             // 1/sqrt(1024)
    float sc[16], m = -3.0e38f;
#pragma unroll
    for (int l = 0; l < 16; ++l) { sc[l] = red[l] * scale; m = fmaxf(m, sc[l]); }
    float den = 0.0f;
#pragma unroll
    for (int l = 0; l < 16; ++l) { sc[l] = __expf(sc[l] - m); den += sc[l]; }
    float inv = 1.0f / den;

    unsigned short* arow = Acat + (size_t)n * K_TOT;
#pragma unroll
    for (int jj = 0; jj < 4; ++jj) {
        float acc = 0.0f;
#pragma unroll
        for (int l = 0; l < 16; ++l) acc += av[jj][l] * sc[l];
        arow[D_IN + H_DIM + j0 + jj] = f2bf(acc * inv);
    }

    const float* xr = x + ((size_t)n * T_STEPS + t) * D_IN;
    for (int d = tid; d < D_IN; d += 256)
        arow[d] = f2bf(xr[d]);
}

// ---------------------------------------------------------------------------
// Fused GEMM + gates, TDM double-buffered pipeline.
// Block: 32 batch rows x 64 hidden units x 4 gates (256 packed cols), 8 waves.
// Per chunk: wave0 issues 2 tensor_load_to_lds (A 32x32, B 256x32) for c+1,
// s_wait_tensorcnt 2, barrier, all waves WMMA out of the other LDS buffer.
// ---------------------------------------------------------------------------
__global__ void __launch_bounds__(256)
lstm_step(unsigned short* __restrict__ Acat,          // (N,K_TOT) bf16
          const unsigned short* __restrict__ WcatT,   // (FH,K_TOT) bf16, gate-interleaved
          const float* __restrict__ bias,             // (4H)
          float* __restrict__ c_st,                   // (N,H)
          float* __restrict__ h_st,                   // (N,H)
          float* __restrict__ out,                    // (N,T,H)
          int t) {
    __shared__ __align__(16) unsigned short bStage[2][256 * KC];  // 2 x 16 KB
    __shared__ __align__(16) unsigned short aStage[2][32 * KC];   // 2 x  2 KB
    __shared__ float a_lds[32 * 256];                             // 32 KB recombine

    const int tid  = threadIdx.x;
    const int wave = tid >> 5;
    const int lane = tid & 31;
    const int li   = lane & 15;
    const int msl  = wave >> 2;                       // 0..1
    const int gate = wave & 3;                        // 0..3

    const int rowBase = blockIdx.x * 32;              // batch rows
    const int pcBase  = blockIdx.y * 256;             // packed weight cols

    const int kHalfA = (lane >> 4) << 4;              // 0 / 16 (A frag halves)
    const int kHalfB = (lane >> 4) << 4;              // 0 / 16 (B frag halves)
    const int aRow   = msl * 16 + li;

    f32x8 acc0 = {}, acc1 = {}, acc2 = {}, acc3 = {};

#if HAVE_TDM
    if (wave == 0) {      // prologue: chunk 0 into buffer 0
        tdm_load_2d(WcatT + (size_t)pcBase * K_TOT,
                    (unsigned)(size_t)&bStage[0][0], K_TOT, FH, KC, 256, K_TOT);
        tdm_load_2d(Acat + (size_t)rowBase * K_TOT,
                    (unsigned)(size_t)&aStage[0][0], K_TOT, N_BATCH, KC, 32, K_TOT);
    }
#endif

    for (int c = 0; c < NCHUNK; ++c) {
        const int buf  = c & 1;
        const int nbuf = buf ^ 1;
        const int k0n  = (c + 1) * KC;

#if HAVE_TDM
        if (wave == 0) {
            if (c + 1 < NCHUNK) {
                tdm_load_2d(WcatT + (size_t)pcBase * K_TOT + k0n,
                            (unsigned)(size_t)&bStage[nbuf][0], K_TOT, FH, KC, 256, K_TOT);
                tdm_load_2d(Acat + (size_t)rowBase * K_TOT + k0n,
                            (unsigned)(size_t)&aStage[nbuf][0], K_TOT, N_BATCH, KC, 32, K_TOT);
                __builtin_amdgcn_s_wait_tensorcnt((short)2);   // chunk c done, c+1 in flight
            } else {
                __builtin_amdgcn_s_wait_tensorcnt((short)0);
            }
        }
        __syncthreads();                                       // publish chunk c
#else
        // cooperative synchronous staging fallback
        if (c == 0) {
            const uint4* sb = (const uint4*)(WcatT + (size_t)(pcBase + tid) * K_TOT);
            uint4* db = (uint4*)(&bStage[0][tid * KC]);
            db[0] = sb[0]; db[1] = sb[1]; db[2] = sb[2]; db[3] = sb[3];
            if (tid < 128) {
                const uint4* sa = (const uint4*)(Acat + (size_t)(rowBase + (tid >> 2)) * K_TOT);
                ((uint4*)(&aStage[0][(tid >> 2) * KC]))[tid & 3] = sa[tid & 3];
            }
        }
        if (c + 1 < NCHUNK) {
            const uint4* sb = (const uint4*)(WcatT + (size_t)(pcBase + tid) * K_TOT + k0n);
            uint4* db = (uint4*)(&bStage[nbuf][tid * KC]);
            db[0] = sb[0]; db[1] = sb[1]; db[2] = sb[2]; db[3] = sb[3];
            if (tid < 128) {
                const uint4* sa = (const uint4*)(Acat + (size_t)(rowBase + (tid >> 2)) * K_TOT + k0n);
                ((uint4*)(&aStage[nbuf][(tid >> 2) * KC]))[tid & 3] = sa[tid & 3];
            }
        }
        __syncthreads();
#endif

        // ---- compute chunk c out of LDS ----
        const unsigned short* aB = &aStage[buf][aRow * KC + kHalfA];
        bf16x8 alo = *reinterpret_cast<const bf16x8*>(aB);
        bf16x8 ahi = *reinterpret_cast<const bf16x8*>(aB + 8);
        bf16x16 afrag = __builtin_shufflevector(alo, ahi,
            0,1,2,3,4,5,6,7,8,9,10,11,12,13,14,15);

        // packed col for (jlocal = nt*16 + li, gate): pc = jlocal*4 + gate
        const unsigned short* bB = &bStage[buf][(li * 4 + gate) * KC + kHalfB];
        bf16x16 b0 = *reinterpret_cast<const bf16x16*>(bB);
        bf16x16 b1 = *reinterpret_cast<const bf16x16*>(bB + 16 * 4 * KC);
        bf16x16 b2 = *reinterpret_cast<const bf16x16*>(bB + 32 * 4 * KC);
        bf16x16 b3 = *reinterpret_cast<const bf16x16*>(bB + 48 * 4 * KC);

        acc0 = __builtin_amdgcn_wmma_f32_16x16x32_bf16(false, afrag, false, b0,
                                                       (short)0, acc0, false, false);
        acc1 = __builtin_amdgcn_wmma_f32_16x16x32_bf16(false, afrag, false, b1,
                                                       (short)0, acc1, false, false);
        acc2 = __builtin_amdgcn_wmma_f32_16x16x32_bf16(false, afrag, false, b2,
                                                       (short)0, acc2, false, false);
        acc3 = __builtin_amdgcn_wmma_f32_16x16x32_bf16(false, afrag, false, b3,
                                                       (short)0, acc3, false, false);

        __syncthreads();                  // reads done before buf is overwritten
    }

    // C/D layout: VGPR r, lanes 0-15 -> M=r, lanes 16-31 -> M=r+8; N = lane&15
    const int mBase = msl * 16 + ((lane >> 4) << 3);
    const int cBase = gate * 64 + li;
#pragma unroll
    for (int r = 0; r < 8; ++r) {
        a_lds[(mBase + r) * 256 + cBase +  0] = acc0[r];
        a_lds[(mBase + r) * 256 + cBase + 16] = acc1[r];
        a_lds[(mBase + r) * 256 + cBase + 32] = acc2[r];
        a_lds[(mBase + r) * 256 + cBase + 48] = acc3[r];
    }
    __syncthreads();

    // gate recombination: 32x64 = 2048 outputs, 8 per thread
#pragma unroll
    for (int it = 0; it < 8; ++it) {
        int idx = tid + it * 256;
        int row = idx >> 6;                    // 0..31
        int col = idx & 63;                    // 0..63
        int n   = blockIdx.x * 32 + row;
        int jg  = blockIdx.y * 64 + col;

        float ai = a_lds[row * 256 +   0 + col] + bias[jg];
        float af = a_lds[row * 256 +  64 + col] + bias[H_DIM + jg];
        float ao = a_lds[row * 256 + 128 + col] + bias[2 * H_DIM + jg];
        float ag = a_lds[row * 256 + 192 + col] + bias[3 * H_DIM + jg];

        float iv = sigmoidf_(ai);
        float fv = sigmoidf_(af);
        float ov = sigmoidf_(ao);
        float gv = tanhf(ag);

        size_t sidx = (size_t)n * H_DIM + jg;
        float cn = fv * c_st[sidx] + iv * gv;
        float hn = ov * tanhf(cn);

        c_st[sidx] = cn;
        h_st[sidx] = hn;
        Acat[(size_t)n * K_TOT + D_IN + jg] = f2bf(hn);
        out[((size_t)n * T_STEPS + t) * H_DIM + jg] = hn;
    }
}

// ---------------------------------------------------------------------------
extern "C" void kernel_launch(void* const* d_in, const int* in_sizes, int n_in,
                              void* d_out, int out_size, void* d_ws, size_t ws_size,
                              hipStream_t stream) {
    const float* x  = (const float*)d_in[0];   // (N,T,D)
    const float* A  = (const float*)d_in[1];   // (N,H,4,4)
    const float* Wx = (const float*)d_in[2];   // (D,4H)
    const float* Wh = (const float*)d_in[3];   // (H,4H)
    const float* Wa = (const float*)d_in[4];   // (H,4H)
    const float* b  = (const float*)d_in[5];   // (4H)
    float* out = (float*)d_out;                // (N,T,H)

    char* ws = (char*)d_ws;
    unsigned short* WcatT = (unsigned short*)ws;                        // 20,971,520 B
    unsigned short* Acat  = (unsigned short*)(ws + 20971520);           //  5,242,880 B
    float*          c_st  = (float*)(ws + 20971520 + 5242880);          //  4,194,304 B
    float*          h_st  = (float*)(ws + 20971520 + 5242880 + 4194304);//  4,194,304 B

    pack_weights<<<FH / 256, 256, 0, stream>>>(Wx, Wh, Wa, WcatT);
    init_state<<<(N_BATCH * H_DIM) / 256, 256, 0, stream>>>(A, c_st, h_st, Acat);

    for (int t = 0; t < T_STEPS; ++t) {
        attn_step<<<N_BATCH, 256, 0, stream>>>(x, A, h_st, Acat, t);
        lstm_step<<<dim3(N_BATCH / 32, H_DIM / 64), 256, 0, stream>>>(
            Acat, WcatT, b, c_st, h_st, out, t);
    }
}
